// GQA_77489799955151
// MI455X (gfx1250) — matrix operand order repaired and checked by reading, not compiled, not run on hardware
//
#include <hip/hip_runtime.h>
#include <hip/hip_bf16.h>

// ---------------- problem constants (from reference) ----------------
constexpr int B_   = 4;
constexpr int S_   = 1024;
constexpr int H_   = 32;
constexpr int HKV_ = 8;
constexpr int D_   = 128;
constexpr int N_   = B_ * S_;            // 4096 tokens
constexpr float SCALE_ = 0.08838834764831845f;   // 1/sqrt(128)
constexpr float NEG_   = -1e9f;

typedef _Float16 h8v  __attribute__((ext_vector_type(8)));
typedef _Float16 h16v __attribute__((ext_vector_type(16)));
typedef float    f8v  __attribute__((ext_vector_type(8)));

__device__ __forceinline__ h16v cat8(h8v a, h8v b) {
    return __builtin_shufflevector(a, b, 0,1,2,3,4,5,6,7,8,9,10,11,12,13,14,15);
}

// -------------------------------------------------------------------
// Kernel 1: RoPE K once -> f16 kr [N,HKV,D]; transpose V -> f16 vT [HKV,D,N].
// K rows are reused by up to 64 query tiles, so RoPE is hoisted out of the
// attention loop.  vT makes the P*V B-operand (keys along WMMA K-dim)
// 16 contiguous halves per lane (two b128 loads, no in-kernel transpose).
// -------------------------------------------------------------------
__global__ __launch_bounds__(256) void rope_pack_kernel(
        const float* __restrict__ k, const float* __restrict__ v,
        const float* __restrict__ theta,
        _Float16* __restrict__ kr, _Float16* __restrict__ vT)
{
    size_t idx = (size_t)blockIdx.x * 256 + threadIdx.x;   // over N*HKV*D
    int    d   = (int)(idx & (D_ - 1));
    int    hk  = (int)((idx >> 7) & (HKV_ - 1));
    size_t n   = idx >> 10;

    float th = theta[n * D_ + d];
    float kv = k[idx];
    size_t pidx = (d < 64) ? idx + 64 : idx - 64;
    float kp = (d < 64) ? -k[pidx] : k[pidx];
    float kro = kv * __cosf(th) + kp * __sinf(th);

    kr[idx] = (_Float16)kro;
    vT[((size_t)hk * D_ + d) * N_ + n] = (_Float16)v[idx];
}

// -------------------------------------------------------------------
// Kernel 2: flash attention.  One wave32 per (b, h, 16-query tile).
// QK^T: 8x v_wmma_f32_16x16x32_f16 per 32-key step (two 16x16 score tiles,
// K=D=128 chained).  Online softmax with cross-lane shfl_xor reductions
// (stay inside each 16-lane half, matching the C-matrix row split).
// P goes C-layout -> A-layout through LDS (96B row stride, 16B-aligned reads).
// P*V: 8x WMMA into a 16x128 f32 accumulator.
// -------------------------------------------------------------------
__global__ __launch_bounds__(32) void gqa_attn_kernel(
        const float* __restrict__ q, const float* __restrict__ theta,
        const _Float16* __restrict__ kr, const _Float16* __restrict__ vT,
        float* __restrict__ o_out, float* __restrict__ l_out)
{
    const int lane = threadIdx.x;         // 0..31
    const int qt   = blockIdx.x;          // 0..63  (16-query tile in segment)
    const int h    = blockIdx.y;          // 0..31
    const int b    = blockIdx.z;          // 0..3
    const int hkv  = h / (H_ / HKV_);
    const int qi0  = qt * 16;             // query start within segment
    const int qn0  = b * S_ + qi0;        // global token start
    const int hf   = lane >> 4;           // which 16-lane half
    const int l16  = lane & 15;

    __shared__ __align__(16) _Float16 pb[16 * 48];   // P staging, stride 96B

    // ---- build RoPE'd Q A-fragments (4 D-slices of 32), ISA A-layout ----
    h16v aq[4];
    {
        const int r = qn0 + l16;                       // this lane's Q row
        const float* qrow = q + ((size_t)r * H_ + h) * D_;
        const float* trow = theta + (size_t)r * D_;
        #pragma unroll
        for (int s = 0; s < 4; ++s) {
            const int kbase = s * 32 + hf * 8;
            #pragma unroll
            for (int i = 0; i < 16; ++i) {
                int d = kbase + i + (i < 8 ? 0 : 8);   // K 0-7 then 16-23 (per half)
                float th = trow[d];
                float qv = qrow[d];
                float qp = (d < 64) ? -qrow[d + 64] : qrow[d - 64];
                aq[s][i] = (_Float16)(qv * __cosf(th) + qp * __sinf(th));
            }
        }
    }

    // ---- accumulators ----
    f8v oacc[8];
    #pragma unroll
    for (int t = 0; t < 8; ++t) oacc[t] = f8v{};
    float mrow[8], lrow[8];
    #pragma unroll
    for (int g = 0; g < 8; ++g) { mrow[g] = -3.0e38f; lrow[g] = 0.0f; }

    const int jend = qi0 + 16;            // causal: keys [0, jend)
    for (int j = 0; j < jend; j += 32) {
        // ---------- QK^T : two 16x16 score tiles ----------
        f8v c1 = f8v{}, c2 = f8v{};
        const int kn1 = b * S_ + j + l16;
        const _Float16* k1p = kr + ((size_t)kn1 * HKV_ + hkv) * D_;
        const _Float16* k2p = k1p + (size_t)16 * HKV_ * D_;
        #pragma unroll
        for (int s = 0; s < 4; ++s) {
            const int d0 = s * 32 + hf * 8;
            h16v bk1 = cat8(*(const h8v*)(k1p + d0), *(const h8v*)(k1p + d0 + 16));
            h16v bk2 = cat8(*(const h8v*)(k2p + d0), *(const h8v*)(k2p + d0 + 16));
            c1 = __builtin_amdgcn_wmma_f32_16x16x32_f16(false, aq[s], false, bk1,
                                                        (short)0, c1, false, false);
            c2 = __builtin_amdgcn_wmma_f32_16x16x32_f16(false, aq[s], false, bk2,
                                                        (short)0, c2, false, false);
        }

        // ---------- scale + causal mask + online softmax ----------
        const int row0 = qi0 + hf * 8;            // C layout: rows row0..row0+7
        const int key1 = j + l16, key2 = key1 + 16;
        float s1[8], s2[8], pm[8];
        #pragma unroll
        for (int g = 0; g < 8; ++g) {
            int row = row0 + g;
            s1[g] = c1[g] * SCALE_ + ((key1 <= row) ? 0.0f : NEG_);
            s2[g] = c2[g] * SCALE_ + ((key2 <= row) ? 0.0f : NEG_);
            pm[g] = fmaxf(s1[g], s2[g]);
        }
        #pragma unroll
        for (int g = 0; g < 8; ++g) {
            #pragma unroll
            for (int off = 1; off < 16; off <<= 1)
                pm[g] = fmaxf(pm[g], __shfl_xor(pm[g], off, 32));
        }
        float alpha[8];
        #pragma unroll
        for (int g = 0; g < 8; ++g) {
            float mn = fmaxf(mrow[g], pm[g]);
            alpha[g] = __expf(mrow[g] - mn);
            mrow[g] = mn;
            s1[g] = __expf(s1[g] - mn);
            s2[g] = __expf(s2[g] - mn);
        }
        float rs[8];
        #pragma unroll
        for (int g = 0; g < 8; ++g) rs[g] = s1[g] + s2[g];
        #pragma unroll
        for (int g = 0; g < 8; ++g) {
            #pragma unroll
            for (int off = 1; off < 16; off <<= 1)
                rs[g] += __shfl_xor(rs[g], off, 32);
            lrow[g] = lrow[g] * alpha[g] + rs[g];
        }
        #pragma unroll
        for (int t = 0; t < 8; ++t)
            #pragma unroll
            for (int g = 0; g < 8; ++g)
                oacc[t][g] *= alpha[g];

        // ---------- P: C-layout -> A-layout via LDS ----------
        #pragma unroll
        for (int g = 0; g < 8; ++g) {
            int row = hf * 8 + g;
            pb[row * 48 + l16]      = (_Float16)s1[g];
            pb[row * 48 + 16 + l16] = (_Float16)s2[g];
        }
        __syncthreads();
        h16v ap;
        {
            int kb = hf * 16;
            ap = cat8(*(const h8v*)(&pb[l16 * 48 + kb]),
                      *(const h8v*)(&pb[l16 * 48 + kb + 8]));
        }
        __syncthreads();

        // ---------- P * V ----------
        const int key0 = b * S_ + j + hf * 16;
        #pragma unroll
        for (int t = 0; t < 8; ++t) {
            int d = t * 16 + l16;
            const _Float16* vp = vT + ((size_t)hkv * D_ + d) * N_ + key0;
            h16v bv = cat8(*(const h8v*)(vp), *(const h8v*)(vp + 8));
            oacc[t] = __builtin_amdgcn_wmma_f32_16x16x32_f16(false, ap, false, bv,
                                                             (short)0, oacc[t], false, false);
        }
    }

    // ---------- epilogue ----------
    float inv[8], lse[8];
    #pragma unroll
    for (int g = 0; g < 8; ++g) {
        inv[g] = 1.0f / lrow[g];
        lse[g] = mrow[g] + __logf(lrow[g]);
    }
    #pragma unroll
    for (int t = 0; t < 8; ++t) {
        #pragma unroll
        for (int g = 0; g < 8; ++g) {
            int r = qn0 + hf * 8 + g;
            int d = t * 16 + l16;
            o_out[((size_t)r * H_ + h) * D_ + d] = oacc[t][g] * inv[g];
        }
    }
    #pragma unroll
    for (int g = 0; g < 8; ++g) {
        if (l16 == g) {
            int r = qn0 + hf * 8 + g;
            l_out[(size_t)r * H_ + h] = lse[g];
        }
    }
}

// -------------------------------------------------------------------
extern "C" void kernel_launch(void* const* d_in, const int* in_sizes, int n_in,
                              void* d_out, int out_size, void* d_ws, size_t ws_size,
                              hipStream_t stream) {
    (void)in_sizes; (void)n_in; (void)out_size; (void)ws_size;
    const float* q     = (const float*)d_in[0];
    const float* k     = (const float*)d_in[1];
    const float* v     = (const float*)d_in[2];
    const float* theta = (const float*)d_in[3];
    // d_in[4] = mask: per-segment causal, applied analytically in-kernel.

    _Float16* kr = (_Float16*)d_ws;                       // [N,HKV,D] f16 (8 MB)
    _Float16* vT = kr + (size_t)N_ * HKV_ * D_;           // [HKV,D,N] f16 (8 MB)

    const int total = N_ * HKV_ * D_;
    rope_pack_kernel<<<total / 256, 256, 0, stream>>>(k, v, theta, kr, vT);

    float* o = (float*)d_out;                             // [N,H,D]
    float* l = o + (size_t)N_ * H_ * D_;                  // [N,H]
    dim3 grid(S_ / 16, H_, B_);
    gqa_attn_kernel<<<grid, 32, 0, stream>>>(q, theta, kr, vT, o, l);
}
